// NNEmb_15857019257246
// MI455X (gfx1250) — compile-verified, matrix operand order checked by reading, not run on hardware
//
#include <hip/hip_runtime.h>

typedef __attribute__((ext_vector_type(16))) __bf16 v16bf;
typedef __attribute__((ext_vector_type(8)))  float  v8f;

#define SEQ   200
#define BATCH 1024
#define DIM   300
#define DPAD  320              // K padded to multiple of 32
#define NTRAIN 100000
#define KSTEPS (DPAD / 32)     // 10 WMMA K-steps

// GEMM tiling: block = 8 waves x 16 rows = 128 M-rows; 32 N-cols staged in LDS
#define MROWS_PER_BLOCK 128
#define STAGE_COLS 32
#define STAGES_PER_BLOCK 25
#define CHUNK_COLS (STAGE_COLS * STAGES_PER_BLOCK)   // 800
#define NXBLK (NTRAIN / CHUNK_COLS)                  // 125
#define ROW_U4 (DPAD * 2 / 16)                       // 40 uint4 per bf16 row
#define STAGE_U4 (STAGE_COLS * ROW_U4)               // 1280 uint4 per stage
#define U4_PER_THREAD (STAGE_U4 / 256)               // 5

#if defined(__has_builtin)
#if __has_builtin(__builtin_amdgcn_global_load_async_to_lds_b128)
#define HAVE_ASYNC_LDS 1
#endif
#endif

// exact parameter type per hipcc diagnostic: vector_size(16) int, AS1 src / AS3 dst
typedef int v4i __attribute__((vector_size(16)));
typedef __attribute__((address_space(1))) v4i g_v4i;
typedef __attribute__((address_space(3))) v4i l_v4i;

// ---- helpers -------------------------------------------------------------

__device__ __forceinline__ unsigned short f2bfu(float f) {   // f32->bf16 RNE
    unsigned u = __float_as_uint(f);
    unsigned r = u + 0x7FFFu + ((u >> 16) & 1u);
    return (unsigned short)(r >> 16);
}
__device__ __forceinline__ unsigned fkey(float f) {          // order-preserving key
    unsigned u = __float_as_uint(f);
    return (u & 0x80000000u) ? ~u : (u | 0x80000000u);
}
__device__ __forceinline__ float fromkey(unsigned k) {
    unsigned u = (k & 0x80000000u) ? (k & 0x7FFFFFFFu) : ~k;
    return __uint_as_float(u);
}
__device__ __forceinline__ void wait_async_zero() {
#if defined(__has_builtin) && __has_builtin(__builtin_amdgcn_s_wait_asynccnt)
    __builtin_amdgcn_s_wait_asynccnt(0);
#else
    asm volatile("s_wait_asynccnt 0x0" ::: "memory");
#endif
}

// ---- kernel 0: init argmax accumulators ----------------------------------

__global__ void nn_init_kernel(unsigned long long* __restrict__ best) {
    int i = blockIdx.x * blockDim.x + threadIdx.x;
    if (i < BATCH) best[i] = 0ull;
}

// ---- kernel 1: embedding gather + mean pool + query norm -----------------

__global__ void __launch_bounds__(DPAD)
nn_pool_kernel(const int* __restrict__ ci, const float* __restrict__ W,
               unsigned short* __restrict__ xb, float* __restrict__ xn_inv) {
    int b = blockIdx.x;
    int d = threadIdx.x;               // 0..319
    float sum = 0.f;
    if (d < DIM) {
        for (int s = 0; s < SEQ; ++s) {
            int tok = ci[s * BATCH + b];               // uniform -> scalar load
            sum += W[(long long)tok * DIM + d];
        }
    }
    float x = sum * (1.0f / (float)SEQ);
    xb[(long long)b * DPAD + d] = (d < DIM) ? f2bfu(x) : (unsigned short)0;

    float ss = (d < DIM) ? x * x : 0.f;
    for (int off = 16; off > 0; off >>= 1) ss += __shfl_xor(ss, off);
    __shared__ float wsum[DPAD / 32];
    if ((threadIdx.x & 31) == 0) wsum[threadIdx.x >> 5] = ss;
    __syncthreads();
    if (threadIdx.x == 0) {
        float t = 0.f;
        for (int i = 0; i < DPAD / 32; ++i) t += wsum[i];
        xn_inv[b] = 1.0f / fmaxf(sqrtf(t), 1e-8f);
    }
}

// ---- kernel 2: train_embs f32 -> bf16 (K-padded, 64 MB -> L2-resident) ---

__global__ void __launch_bounds__(DPAD)
nn_convert_kernel(const float* __restrict__ tr,
                  unsigned short* __restrict__ tb, float* __restrict__ tn_inv) {
    int r = blockIdx.x;
    int d = threadIdx.x;
    float v = (d < DIM) ? tr[(long long)r * DIM + d] : 0.f;
    tb[(long long)r * DPAD + d] = (d < DIM) ? f2bfu(v) : (unsigned short)0;

    float ss = v * v;
    for (int off = 16; off > 0; off >>= 1) ss += __shfl_xor(ss, off);
    __shared__ float wsum[DPAD / 32];
    if ((threadIdx.x & 31) == 0) wsum[threadIdx.x >> 5] = ss;
    __syncthreads();
    if (threadIdx.x == 0) {
        float t = 0.f;
        for (int i = 0; i < DPAD / 32; ++i) t += wsum[i];
        tn_inv[r] = 1.0f / fmaxf(sqrtf(t), 1e-8f);
    }
}

// ---- kernel 3: async-LDS bf16 WMMA GEMM + fused cosine argmax ------------
// grid (NXBLK, BATCH/128). 8 waves each own a 16-row M tile (A in VGPRs);
// 32-col B tiles stream global->LDS via GLOBAL_LOAD_ASYNC_TO_LDS_B128
// (no VGPR staging, ASYNCcnt-tracked), double-buffered against 20 WMMAs.

__global__ void __launch_bounds__(256)
nn_gemm_argmax_kernel(const unsigned short* __restrict__ xb,
                      const unsigned short* __restrict__ tb,
                      const float* __restrict__ xn_inv,
                      const float* __restrict__ tn_inv,
                      unsigned long long* __restrict__ best) {
    __shared__ uint4 sbuf[2][STAGE_U4];                 // 2 x 20 KB

    const int t    = threadIdx.x;
    const int lane = t & 31;
    const int wave = t >> 5;                            // 0..7
    const int ln   = lane & 15;
    const int ksel = lane >> 4;                         // K-half carried by lane
    const int mrow0 = blockIdx.y * MROWS_PER_BLOCK + wave * 16;
    const int nbase = blockIdx.x * CHUNK_COLS;

    const uint4* tv = reinterpret_cast<const uint4*>(tb);
    // this thread's fixed slice of each stage: rows/cols within the 32x320 tile
    int srow[U4_PER_THREAD], scol[U4_PER_THREAD];
#pragma unroll
    for (int j = 0; j < U4_PER_THREAD; ++j) {
        int idx = t + j * 256;                          // 0..1279
        srow[j] = idx / ROW_U4;
        scol[j] = idx % ROW_U4;
    }

    // A fragments: this wave's 16x320 bf16 query tile, resident in VGPRs
    v16bf afrag[KSTEPS];
    const unsigned short* arow = xb + (long long)(mrow0 + ln) * DPAD + ksel * 16;
#pragma unroll
    for (int kt = 0; kt < KSTEPS; ++kt)
        afrag[kt] = *reinterpret_cast<const v16bf*>(arow + kt * 32);

    float xr[8];
#pragma unroll
    for (int j = 0; j < 8; ++j) xr[j] = xn_inv[mrow0 + j + (ksel << 3)];

    unsigned bkey[8], bidx[8];
#pragma unroll
    for (int j = 0; j < 8; ++j) { bkey[j] = 0u; bidx[j] = 0u; }

    // prologue: kick off stage 0 into buffer 0
#if defined(HAVE_ASYNC_LDS)
#pragma unroll
    for (int j = 0; j < U4_PER_THREAD; ++j)
        __builtin_amdgcn_global_load_async_to_lds_b128(
            (g_v4i*)(tv + (size_t)(nbase + srow[j]) * ROW_U4 + scol[j]),
            (l_v4i*)&sbuf[0][t + j * 256], 0, 0);
    wait_async_zero();
#else
#pragma unroll
    for (int j = 0; j < U4_PER_THREAD; ++j)
        sbuf[0][t + j * 256] = tv[(size_t)(nbase + srow[j]) * ROW_U4 + scol[j]];
#endif
    __syncthreads();

    for (int s = 0; s < STAGES_PER_BLOCK; ++s) {
        const int pb = s & 1;
        const int n0 = nbase + s * STAGE_COLS;

#if defined(HAVE_ASYNC_LDS)
        // async engine fills the other buffer while we run 20 WMMAs
        if (s + 1 < STAGES_PER_BLOCK) {
            const int nn = nbase + (s + 1) * STAGE_COLS;
#pragma unroll
            for (int j = 0; j < U4_PER_THREAD; ++j)
                __builtin_amdgcn_global_load_async_to_lds_b128(
                    (g_v4i*)(tv + (size_t)(nn + srow[j]) * ROW_U4 + scol[j]),
                    (l_v4i*)&sbuf[pb ^ 1][t + j * 256], 0, 0);
        }
#endif

        // two 16-col sub-tiles share each A fragment -> 2 independent chains
        const char* bbase = reinterpret_cast<const char*>(&sbuf[pb][0]);
        v8f c0 = {}, c1 = {};
#pragma unroll
        for (int kt = 0; kt < KSTEPS; ++kt) {
            v16bf b0 = *reinterpret_cast<const v16bf*>(
                bbase + (size_t)ln * (DPAD * 2) + kt * 64 + ksel * 32);
            v16bf b1 = *reinterpret_cast<const v16bf*>(
                bbase + (size_t)(16 + ln) * (DPAD * 2) + kt * 64 + ksel * 32);
            c0 = __builtin_amdgcn_wmma_f32_16x16x32_bf16(
                     false, afrag[kt], false, b0, (short)0, c0, false, false);
            c1 = __builtin_amdgcn_wmma_f32_16x16x32_bf16(
                     false, afrag[kt], false, b1, (short)0, c1, false, false);
        }

        // fused cosine scale + running argmax (sub-tile 0 then 1)
        const float ts0 = tn_inv[n0 + ln];
        const float ts1 = tn_inv[n0 + 16 + ln];
        const unsigned ni0 = (unsigned)(n0 + ln);
        const unsigned ni1 = (unsigned)(n0 + 16 + ln);
#pragma unroll
        for (int j = 0; j < 8; ++j) {
            float s0 = c0[j] * ts0 * xr[j];
            unsigned k0 = fkey(s0);
            if (k0 > bkey[j]) { bkey[j] = k0; bidx[j] = ni0; }
            float s1 = c1[j] * ts1 * xr[j];
            unsigned k1 = fkey(s1);
            if (k1 > bkey[j]) { bkey[j] = k1; bidx[j] = ni1; }
        }

#if defined(HAVE_ASYNC_LDS)
        if (s + 1 < STAGES_PER_BLOCK) wait_async_zero();   // next buffer landed
        __syncthreads();
#else
        __syncthreads();
        if (s + 1 < STAGES_PER_BLOCK) {
            const int nn = nbase + (s + 1) * STAGE_COLS;
#pragma unroll
            for (int j = 0; j < U4_PER_THREAD; ++j)        // short live ranges
                sbuf[pb ^ 1][t + j * 256] =
                    tv[(size_t)(nn + srow[j]) * ROW_U4 + scol[j]];
            __syncthreads();
        }
#endif
    }

    // argmax reduce across the 16 lanes sharing each C row, merge globally
#pragma unroll
    for (int j = 0; j < 8; ++j) {
        unsigned key = bkey[j], idx = bidx[j];
        for (int off = 8; off > 0; off >>= 1) {
            unsigned ok = __shfl_xor(key, off);
            unsigned oi = __shfl_xor(idx, off);
            if (ok > key || (ok == key && oi > idx)) { key = ok; idx = oi; }
        }
        if (ln == 0) {
            int m = mrow0 + j + (ksel << 3);
            unsigned long long kv = ((unsigned long long)key << 32) | (unsigned long long)idx;
            atomicMax(&best[m], kv);
        }
    }
}

// ---- kernel 4: unpack argmax, gather labels ------------------------------

__global__ void nn_finalize_kernel(const unsigned long long* __restrict__ best,
                                   const int* __restrict__ y_train,
                                   float* __restrict__ out) {
    int b = blockIdx.x * blockDim.x + threadIdx.x;
    if (b < BATCH) {
        unsigned long long kv = best[b];
        unsigned key = (unsigned)(kv >> 32);
        int idx = (int)(kv & 0xFFFFFFFFull);
        out[b]         = (float)y_train[idx];   // y_pred
        out[BATCH + b] = fromkey(key);          // scores
    }
}

// ---- launcher ------------------------------------------------------------

extern "C" void kernel_launch(void* const* d_in, const int* in_sizes, int n_in,
                              void* d_out, int out_size, void* d_ws, size_t ws_size,
                              hipStream_t stream) {
    const int*   ci = (const int*)d_in[0];      // [200, 1024]
    const float* W  = (const float*)d_in[1];    // [50000, 300]
    const float* tr = (const float*)d_in[2];    // [100000, 300]
    const int*   y  = (const int*)d_in[3];      // [100000]
    float* out = (float*)d_out;                 // [1024 y_pred | 1024 scores]

    char* ws = (char*)d_ws;
    size_t off = 0;
    unsigned short* xb = (unsigned short*)(ws + off); off += (size_t)BATCH * DPAD * 2;
    unsigned short* tb = (unsigned short*)(ws + off); off += (size_t)NTRAIN * DPAD * 2;
    float* xn_inv = (float*)(ws + off); off += (size_t)BATCH * 4 + 3072;
    float* tn_inv = (float*)(ws + off); off += (size_t)NTRAIN * 4;
    unsigned long long* best = (unsigned long long*)(ws + off);

    nn_init_kernel<<<dim3((BATCH + 255) / 256), dim3(256), 0, stream>>>(best);
    nn_pool_kernel<<<dim3(BATCH), dim3(DPAD), 0, stream>>>(ci, W, xb, xn_inv);
    nn_convert_kernel<<<dim3(NTRAIN), dim3(DPAD), 0, stream>>>(tr, tb, tn_inv);
    nn_gemm_argmax_kernel<<<dim3(NXBLK, BATCH / MROWS_PER_BLOCK), dim3(256), 0, stream>>>(
        xb, tb, xn_inv, tn_inv, best);
    nn_finalize_kernel<<<dim3((BATCH + 255) / 256), dim3(256), 0, stream>>>(best, y, out);
}
